// Multihead_2894807957733
// MI455X (gfx1250) — matrix operand order repaired and checked by reading, not compiled, not run on hardware
//
#include <hip/hip_runtime.h>
#include <hip/hip_bf16.h>

#define BB 4
#define SS 2048
#define DD 1024
#define HH 16
#define HD 64
#define HP 32

typedef __bf16 bf16;
typedef bf16  v4bf  __attribute__((ext_vector_type(4)));
typedef bf16  v8bf  __attribute__((ext_vector_type(8)));
typedef bf16  v16bf __attribute__((ext_vector_type(16)));
typedef float v8f   __attribute__((ext_vector_type(8)));

#if defined(__has_builtin)
#if __has_builtin(__builtin_amdgcn_global_load_async_to_lds_b128) && \
    __has_builtin(__builtin_amdgcn_s_wait_asynccnt)
#define HAS_ASYNC 1
#endif
#endif
#ifndef HAS_ASYNC
#define HAS_ASYNC 0
#endif

#if HAS_ASYNC
typedef int v4i_ __attribute__((vector_size(16)));
typedef __attribute__((address_space(1))) v4i_* as1_v4i;
typedef __attribute__((address_space(3))) v4i_* as3_v4i;
__device__ __forceinline__ void async_b128(const void* g, void* l) {
  __builtin_amdgcn_global_load_async_to_lds_b128((as1_v4i)g, (as3_v4i)l, 0, 0);
}
__device__ __forceinline__ void wait_async0() { __builtin_amdgcn_s_wait_asynccnt(0); }
#else
__device__ __forceinline__ void wait_async0() {}
#endif

__device__ __forceinline__ v8f wmma_bf16(v16bf a, v16bf b, v8f c) {
  return __builtin_amdgcn_wmma_f32_16x16x32_bf16(false, a, false, b, (short)0, c, false, false);
}

__device__ __forceinline__ v16bf cat8(v8bf lo, v8bf hi) {
  v16bf r;
#pragma unroll
  for (int i = 0; i < 8; ++i) { r[i] = lo[i]; r[i + 8] = hi[i]; }
  return r;
}

// build a 16-element bf16 fragment from two 8-float LDS chunks
__device__ __forceinline__ v16bf cvt16(const float* p0, const float* p1) {
  v16bf r;
#pragma unroll
  for (int i = 0; i < 8; ++i) { r[i] = (bf16)p0[i]; r[i + 8] = (bf16)p1[i]; }
  return r;
}

// ---------------------------------------------------------------------------
// C[M,N] = A[M,K] @ W[N,K]^T + bias ; f32 in/out, bf16 WMMA compute.
// f32 tiles double-buffered in LDS via async-to-LDS copies; convert to bf16
// at fragment-build time. Block 256 thr = 8 waves; 128x128 tile.
// ---------------------------------------------------------------------------
#define GBM 128
#define GBN 128
#define GBK 32

__global__ __launch_bounds__(256) void gemm_xwt(
    const float* __restrict__ A, const float* __restrict__ W,
    const float* __restrict__ bias, float* __restrict__ C,
    int M, int N, int K)
{
  __shared__ float As[2][GBM][GBK];   // 2 x 16 KB
  __shared__ float Bs[2][GBN][GBK];   // 2 x 16 KB
  const int tid  = threadIdx.x;
  const int lane = tid & 31;
  const int wave = tid >> 5;
  const int half = lane >> 4;
  const int lr   = lane & 15;
  const int wm   = wave >> 1;
  const int wn   = wave & 1;
  const int bm   = blockIdx.x * GBM;
  const int bn   = blockIdx.y * GBN;

  auto issueTile = [&](int buf, int k0) {
#if HAS_ASYNC
#pragma unroll
    for (int i = 0; i < 4; ++i) {
      int e = (tid + i * 256) * 4;          // float index in 128x32 tile
      int r = e >> 5, c = e & 31;
      async_b128(A + (size_t)(bm + r) * K + k0 + c, &As[buf][r][c]);
      async_b128(W + (size_t)(bn + r) * K + k0 + c, &Bs[buf][r][c]);
    }
#else
    float4 fa[4], fw[4];
#pragma unroll
    for (int i = 0; i < 4; ++i) {           // batch all loads first
      int e = (tid + i * 256) * 4;
      int r = e >> 5, c = e & 31;
      fa[i] = *(const float4*)(A + (size_t)(bm + r) * K + k0 + c);
      fw[i] = *(const float4*)(W + (size_t)(bn + r) * K + k0 + c);
    }
#pragma unroll
    for (int i = 0; i < 4; ++i) {
      int e = (tid + i * 256) * 4;
      int r = e >> 5, c = e & 31;
      *(float4*)&As[buf][r][c] = fa[i];
      *(float4*)&Bs[buf][r][c] = fw[i];
    }
    if (k0 + GBK < K) {
      __builtin_prefetch(A + (size_t)(bm + (tid >> 1)) * K + k0 + GBK + (tid & 1) * 16, 0, 1);
      __builtin_prefetch(W + (size_t)(bn + (tid >> 1)) * K + k0 + GBK + (tid & 1) * 16, 0, 1);
    }
#endif
  };

  v8f acc[2][4];
#pragma unroll
  for (int s = 0; s < 2; ++s)
#pragma unroll
    for (int j = 0; j < 4; ++j) acc[s][j] = 0.0f;

  issueTile(0, 0);
  wait_async0();
  __syncthreads();

  for (int k0 = 0; k0 < K; k0 += GBK) {
    const int cur = (k0 / GBK) & 1, nxt = cur ^ 1;
    if (k0 + GBK < K) issueTile(nxt, k0 + GBK);   // overlap copy with compute

    v16bf af[2];
#pragma unroll
    for (int s = 0; s < 2; ++s) {
      int row = wm * 32 + s * 16 + lr;
      af[s] = cvt16(&As[cur][row][8 * half], &As[cur][row][16 + 8 * half]);
    }
#pragma unroll
    for (int j = 0; j < 4; ++j) {
      int col = wn * 64 + j * 16 + lr;
      v16bf bfr = cvt16(&Bs[cur][col][8 * half], &Bs[cur][col][16 + 8 * half]);
#pragma unroll
      for (int s = 0; s < 2; ++s) acc[s][j] = wmma_bf16(af[s], bfr, acc[s][j]);
    }

    wait_async0();
    __syncthreads();
  }

#pragma unroll
  for (int s = 0; s < 2; ++s)
#pragma unroll
    for (int j = 0; j < 4; ++j) {
      int col = bn + wn * 64 + j * 16 + lr;
      float bvv = bias ? bias[col] : 0.0f;
#pragma unroll
      for (int r = 0; r < 8; ++r) {
        int row = bm + wm * 32 + s * 16 + r + 8 * half;
        C[(size_t)row * N + col] = acc[s][j][r] + bvv;
      }
    }
}

// ---------------------------------------------------------------------------
// RoPE + scale + f32->bf16 repack:  [B,S,H,HD] f32 -> [B,H,S,HD] bf16
// ---------------------------------------------------------------------------
__global__ void rope_pack(const float* __restrict__ Xf, const float* __restrict__ inv_freq,
                          const float* __restrict__ rbias, bf16* __restrict__ Xr, float scale)
{
  int idx = blockIdx.x * blockDim.x + threadIdx.x;
  const int total = BB * SS * HH * HP;
  if (idx >= total) return;
  int i = idx % HP;
  int h = (idx / HP) % HH;
  int t = (idx / (HP * HH)) % SS;
  int b = idx / (HP * HH * SS);
  float f = (float)t * inv_freq[i] + rbias[t * HP + i];
  float c = __cosf(f), s = __sinf(f);
  size_t src = (((size_t)b * SS + t) * HH + h) * HD + 2 * i;
  float xr = Xf[src], xi = Xf[src + 1];
  float yr = (xr * c - xi * s) * scale;
  float yi = (xr * s + xi * c) * scale;
  size_t dst = (((size_t)b * HH + h) * SS + t) * HD + 2 * i;
  Xr[dst]     = (bf16)yr;
  Xr[dst + 1] = (bf16)yi;
}

// ---------------------------------------------------------------------------
// V repack d-major: [B,S,H,HD] f32 -> [B,H,HD,S] bf16
// ---------------------------------------------------------------------------
__global__ void pack_v(const float* __restrict__ Vf, bf16* __restrict__ Vt)
{
  int idx = blockIdx.x * blockDim.x + threadIdx.x;
  const int total = BB * SS * HH * HD;
  if (idx >= total) return;
  int d = idx % HD;
  int h = (idx / HD) % HH;
  int t = (idx / (HD * HH)) % SS;
  int b = idx / (HD * HH * SS);
  size_t dst = (((size_t)b * HH + h) * HD + d) * SS + t;
  Vt[dst] = (bf16)Vf[idx];
}

// ---------------------------------------------------------------------------
// Flash attention; also emits pre-softmax qk (+mask) to d_out.
// Block: one (b,h), 128 q rows, 8 waves x 16 rows; key tiles of 32,
// double-buffered in LDS via async-to-LDS copies.
// ---------------------------------------------------------------------------
__global__ __launch_bounds__(256) void attention(
    const bf16* __restrict__ Qr, const bf16* __restrict__ Kr, const bf16* __restrict__ Vt,
    const float* __restrict__ mask, float* __restrict__ qk, float* __restrict__ attn)
{
  __shared__ bf16 Ks[2][32][64];    // keys tile, row-major in d  (2 x 4 KB)
  __shared__ bf16 Vs[2][64][32];    // V tile, d-major            (2 x 4 KB)
  __shared__ bf16 Ps[8][16][32];    // per-wave P staging (C-layout -> A-layout)

  const int tid  = threadIdx.x;
  const int lane = tid & 31;
  const int wave = tid >> 5;
  const int half = lane >> 4;
  const int lr   = lane & 15;
  const int bh   = blockIdx.y;
  const int b    = bh / HH, h = bh % HH;
  const int q0   = blockIdx.x * 128 + wave * 16;

  const bf16* Kb = Kr + (size_t)bh * SS * HD;
  const bf16* Vb = Vt + (size_t)bh * HD * SS;

  auto loadKV = [&](int buf, int k0) {
    int d = tid >> 2, c = (tid & 3) * 8;
#if HAS_ASYNC
    async_b128(Kb + (size_t)k0 * HD + tid * 8, &Ks[buf][0][0] + tid * 8);
    async_b128(Vb + (size_t)d * SS + k0 + c, &Vs[buf][d][c]);
#else
    v8bf kv = *(const v8bf*)(Kb + (size_t)k0 * HD + tid * 8);
    v8bf vv = *(const v8bf*)(Vb + (size_t)d * SS + k0 + c);
    *(v8bf*)(&Ks[buf][0][0] + tid * 8) = kv;
    *(v8bf*)&Vs[buf][d][c] = vv;
#endif
  };

  // Q fragments (16 rows x 64 d) loaded once from global
  const bf16* Qp = Qr + ((size_t)bh * SS + q0) * HD;
  v16bf qf[2];
#pragma unroll
  for (int dt = 0; dt < 2; ++dt) {
    v8bf lo = *(const v8bf*)(Qp + (size_t)lr * HD + dt * 32 + 8 * half);
    v8bf hi = *(const v8bf*)(Qp + (size_t)lr * HD + dt * 32 + 16 + 8 * half);
    qf[dt] = cat8(lo, hi);
  }

  v8f m_run, l_run, O[4];
#pragma unroll
  for (int r = 0; r < 8; ++r) { m_run[r] = -3.0e38f; l_run[r] = 0.0f; }
#pragma unroll
  for (int j = 0; j < 4; ++j) O[j] = 0.0f;

  const size_t qkbase = (size_t)bh * SS * SS;

  loadKV(0, 0);
  wait_async0();
  __syncthreads();

  for (int k0 = 0; k0 < SS; k0 += 32) {
    const int cur = (k0 >> 5) & 1, nxt = cur ^ 1;
    if (k0 + 32 < SS) loadKV(nxt, k0 + 32);      // overlap copy with compute

    // S = Q . K^T  (two 16x16 key tiles, K-dim = 64 via two k32 WMMAs)
    v8f sf[2];
    sf[0] = 0.0f; sf[1] = 0.0f;
#pragma unroll
    for (int nt = 0; nt < 2; ++nt)
#pragma unroll
      for (int dt = 0; dt < 2; ++dt) {
        v16bf kb = cat8(*(const v8bf*)&Ks[cur][nt * 16 + lr][dt * 32 + 8 * half], *(const v8bf*)&Ks[cur][nt * 16 + lr][dt * 32 + 16 + 8 * half]);
        sf[nt] = wmma_bf16(qf[dt], kb, sf[nt]);
      }

    // mask add, emit qk, online softmax
    float pv[2][8];
    v8f tmax;
#pragma unroll
    for (int r = 0; r < 8; ++r) tmax[r] = -3.0e38f;
#pragma unroll
    for (int nt = 0; nt < 2; ++nt)
#pragma unroll
      for (int r = 0; r < 8; ++r) {
        int qg = q0 + r + 8 * half;
        int kg = k0 + nt * 16 + lr;
        float val = sf[nt][r] + mask[(size_t)qg * SS + kg];
        qk[qkbase + (size_t)qg * SS + kg] = val;
        pv[nt][r] = val;
        tmax[r] = fmaxf(tmax[r], val);
      }
#pragma unroll
    for (int off = 8; off >= 1; off >>= 1)
#pragma unroll
      for (int r = 0; r < 8; ++r) tmax[r] = fmaxf(tmax[r], __shfl_xor(tmax[r], off, 32));

    v8f sc, psum;
#pragma unroll
    for (int r = 0; r < 8; ++r) {
      float mn = fmaxf(m_run[r], tmax[r]);
      sc[r] = __expf(m_run[r] - mn);
      m_run[r] = mn;
      psum[r] = 0.0f;
    }
#pragma unroll
    for (int nt = 0; nt < 2; ++nt)
#pragma unroll
      for (int r = 0; r < 8; ++r) {
        float p = __expf(pv[nt][r] - m_run[r]);
        psum[r] += p;
        Ps[wave][r + 8 * half][nt * 16 + lr] = (bf16)p;
      }
#pragma unroll
    for (int off = 8; off >= 1; off >>= 1)
#pragma unroll
      for (int r = 0; r < 8; ++r) psum[r] += __shfl_xor(psum[r], off, 32);
#pragma unroll
    for (int r = 0; r < 8; ++r) l_run[r] = l_run[r] * sc[r] + psum[r];
#pragma unroll
    for (int j = 0; j < 4; ++j)
#pragma unroll
      for (int r = 0; r < 8; ++r) O[j][r] *= sc[r];

    // P as A-fragment (LDS round-trip, per-wave private -> wave-ordered DS)
    v8bf plo = *(const v8bf*)&Ps[wave][lr][8 * half];
    v8bf phi = *(const v8bf*)&Ps[wave][lr][16 + 8 * half];
    v16bf pa = cat8(plo, phi);
#pragma unroll
    for (int j = 0; j < 4; ++j) {
      v16bf vbf = cat8(*(const v8bf*)&Vs[cur][j * 16 + lr][8 * half], *(const v8bf*)&Vs[cur][j * 16 + lr][16 + 8 * half]);
      O[j] = wmma_bf16(pa, vbf, O[j]);
    }

    wait_async0();
    __syncthreads();
  }

#pragma unroll
  for (int j = 0; j < 4; ++j)
#pragma unroll
    for (int r = 0; r < 8; ++r) {
      int qg = q0 + r + 8 * half;
      attn[((size_t)b * SS + qg) * DD + h * HD + j * 16 + lr] = O[j][r] / l_run[r];
    }
}

// ---------------------------------------------------------------------------
extern "C" void kernel_launch(void* const* d_in, const int* in_sizes, int n_in,
                              void* d_out, int out_size, void* d_ws, size_t ws_size,
                              hipStream_t stream)
{
  const float* x        = (const float*)d_in[0];
  const float* Wq       = (const float*)d_in[1];
  const float* bq       = (const float*)d_in[2];
  const float* Wk       = (const float*)d_in[3];
  const float* Wv       = (const float*)d_in[4];
  const float* bv       = (const float*)d_in[5];
  const float* Wo       = (const float*)d_in[6];
  const float* bo       = (const float*)d_in[7];
  const float* inv_freq = (const float*)d_in[8];
  const float* rbias    = (const float*)d_in[9];
  const float* mask     = (const float*)d_in[10];

  float* out = (float*)d_out;
  float* qk  = out + (size_t)BB * SS * DD;

  char* ws = (char*)d_ws;
  const size_t MSD = (size_t)BB * SS * DD * sizeof(float);   // 33.5 MB
  float* Qf = (float*)(ws);                 // later reused as attention output
  float* Kf = (float*)(ws + MSD);
  float* Vf = (float*)(ws + 2 * MSD);
  bf16*  Qr = (bf16*)(ws + 3 * MSD);
  bf16*  Kr = (bf16*)(ws + 3 * MSD + MSD / 2);
  bf16*  Vt = (bf16*)(ws + 4 * MSD);

  const int M = BB * SS, N = DD, K = DD;
  dim3 ggrid(M / GBM, N / GBN);
  gemm_xwt<<<ggrid, 256, 0, stream>>>(x, Wq, bq,      Qf, M, N, K);
  gemm_xwt<<<ggrid, 256, 0, stream>>>(x, Wk, nullptr, Kf, M, N, K);
  gemm_xwt<<<ggrid, 256, 0, stream>>>(x, Wv, bv,      Vf, M, N, K);

  const float scale = 0.35355339059327373f;   // 64^-0.25
  int pairs = BB * SS * HH * HP;
  rope_pack<<<(pairs + 255) / 256, 256, 0, stream>>>(Qf, inv_freq, rbias, Qr, scale);
  rope_pack<<<(pairs + 255) / 256, 256, 0, stream>>>(Kf, inv_freq, rbias, Kr, scale);
  int velems = BB * SS * HH * HD;
  pack_v<<<(velems + 255) / 256, 256, 0, stream>>>(Vf, Vt);

  dim3 agrid(SS / 128, BB * HH);
  attention<<<agrid, 256, 0, stream>>>(Qr, Kr, Vt, mask, qk, Qf);

  gemm_xwt<<<ggrid, 256, 0, stream>>>(Qf, Wo, bo, out, M, N, K);
}
